// MultiHeadAttention_7275674600650
// MI455X (gfx1250) — matrix operand order repaired and checked by reading, not compiled, not run on hardware
//
#include <hip/hip_runtime.h>
#include <hip/hip_bf16.h>
#include <hip/hip_fp16.h>
#include <cstdint>

// ---------------------------------------------------------------------------
// MI455X (gfx1250) fused MHA forward.
//  - f16 WMMA (V_WMMA_F32_16X16X32_F16), fp32 accumulate
//  - wave32, 128-thread blocks (4 waves); 128x64 GEMM block tiles
//  - RoPE fused in-register in the QKV epilogue
//  - flash attention with TDM (tensor_load_to_lds) double-buffered K/V tiles
// ---------------------------------------------------------------------------

typedef _Float16 h16;
typedef unsigned int u32;
typedef __attribute__((ext_vector_type(16))) _Float16 v16h;
typedef __attribute__((ext_vector_type(8)))  _Float16 v8h;
typedef __attribute__((ext_vector_type(8)))  float    v8f;
typedef __attribute__((ext_vector_type(4)))  u32      u32x4;
typedef __attribute__((ext_vector_type(8)))  int      i32x8;
typedef __attribute__((ext_vector_type(4)))  int      i32x4;

static constexpr int NB     = 2;
static constexpr int SEQ    = 2048;
static constexpr int HDIM   = 1024;
static constexpr int NHEADS = 16;
static constexpr int HSZ    = 64;     // head dim
static constexpr int LDA    = 40;     // LDS row stride (halves), 32-wide k tiles (80B)
static constexpr int LDK    = 72;     // LDS row stride (halves), 64-wide tiles (144B)

#define WMMA_F16(a, b, c) \
  __builtin_amdgcn_wmma_f32_16x16x32_f16(false, (a), false, (b), (short)0, (c), false, false)

#if __has_builtin(__builtin_amdgcn_tensor_load_to_lds) && \
    __has_builtin(__builtin_amdgcn_s_wait_tensorcnt)
#define USE_TDM 1
#else
#define USE_TDM 0
#endif

// A-matrix fragment (16x32 f16): lane L -> row = row0 + (L&15);
// k chunks at kBase + (L<16?0:8) and +16 (ISA 7.12.2, 16-bit A layout).
__device__ __forceinline__ v16h frag_a(const h16* __restrict__ base, int row0,
                                       int stride, int kBase) {
  const int lane = threadIdx.x & 31;
  const h16* p = base + (size_t)(row0 + (lane & 15)) * stride + kBase + ((lane >> 4) << 3);
  v8h lo = *(const v8h*)p;
  v8h hi = *(const v8h*)(p + 16);
  v16h r;
#pragma unroll
  for (int i = 0; i < 8; ++i) { r[i] = lo[i]; r[i + 8] = hi[i]; }
  return r;
}

// B-matrix fragment (32x16 f16), source "row n contiguous along k":
// lane L -> column n = row0 + (L&15); k chunks at kBase + (L<16?0:8) and +16.
__device__ __forceinline__ v16h frag_b(const h16* __restrict__ base, int row0,
                                       int stride, int kBase) {
  const int lane = threadIdx.x & 31;
  const h16* p = base + (size_t)(row0 + (lane & 15)) * stride + kBase + ((lane >> 4) << 3);
  v8h lo = *(const v8h*)p;
  v8h hi = *(const v8h*)(p + 16);
  v16h r;
#pragma unroll
  for (int i = 0; i < 8; ++i) { r[i] = lo[i]; r[i + 8] = hi[i]; }
  return r;
}

__device__ __forceinline__ v8h cvt8(float4 a, float4 b) {
  v8h r = {(h16)a.x, (h16)a.y, (h16)a.z, (h16)a.w,
           (h16)b.x, (h16)b.y, (h16)b.z, (h16)b.w};
  return r;
}

#if USE_TDM
__device__ __forceinline__ u32 ldsOffset(const void* p) {
  // generic LDS address = {aperture, offset[31:0]}  -> low 32 bits are the offset
  return (u32)(uintptr_t)p;
}

// 2D TDM tile load, 2-byte elements, LDS padding: +4 dwords every 128B
// (pad_interval=4, pad_amount=3) -> 64-half rows land at stride LDK=72.
__device__ __forceinline__ void tdm_load_2d(u32 ldsOff, const void* gaddr,
                                            u32 tensor_d0, u32 tensor_d1,
                                            u32 tile_d0, u32 tile_d1, u32 stride0) {
  const uint64_t ga = (uint64_t)(uintptr_t)gaddr;
  u32x4 g0;
  g0[0] = 1u;                                   // count=1, user mode
  g0[1] = ldsOff;                               // lds_addr
  g0[2] = (u32)ga;                              // global_addr[31:0]
  g0[3] = (u32)((ga >> 32) & 0x1ffffffu) | (2u << 30);  // global_addr[56:32] | type=2
  i32x8 g1;
  g1[0] = (int)((1u << 16) | (1u << 20) | (4u << 22) | (3u << 25));
  //            data_size=2B  pad_enable   pad_ivl=128B  pad_amt=4dw
  g1[1] = (int)(tensor_d0 << 16);                          // tensor_dim0[15:0]
  g1[2] = (int)((tensor_d0 >> 16) | (tensor_d1 << 16));    // td0[31:16] | td1[15:0]
  g1[3] = (int)((tensor_d1 >> 16) | (tile_d0 << 16));      // td1[31:16] | tile_dim0
  g1[4] = (int)(tile_d1 & 0xffffu);                        // tile_dim1 | tile_dim2=0
  g1[5] = (int)stride0;                                    // tensor_dim0_stride[31:0]
  g1[6] = 0;                                               // stride0 hi | stride1 lo
  g1[7] = 0;
  const i32x4 z4 = {0, 0, 0, 0};
  const i32x8 z8 = {0, 0, 0, 0, 0, 0, 0, 0};
  // 6-arg form (clang-23 / therock-10.0 headers)
  __builtin_amdgcn_tensor_load_to_lds(g0, g1, z4, z4, z8, 0);
}
#endif

// ---------------------------------------------------------------------------
// Kernel 1: QKV projection (+ fused RoPE for Q/K, transpose-store for V).
// grid = (B*S/128, NHEADS, 3), block = 128.  Block tile: 128(M) x 64(N).
// Q/K ws: [B][NH][S][HD] f16 (RoPE applied).  V ws: [B][NH][HD][S] f16.
// ---------------------------------------------------------------------------
__global__ __launch_bounds__(128)
void qkv_rope_kernel(const float* __restrict__ X,
                     const float* __restrict__ Wq,
                     const float* __restrict__ Wk,
                     const float* __restrict__ Wv,
                     h16* __restrict__ Qws, h16* __restrict__ Kws,
                     h16* __restrict__ Vtws) {
  __shared__ h16   Xs[128 * LDA];
  __shared__ h16   Ws[64 * LDA];
  __shared__ float stage[128 * 65];

  const int t     = threadIdx.x;
  const int wave  = t >> 5;
  const int lane  = t & 31;
  const int mBase = blockIdx.x * 128;         // row in flattened [B*S, H]
  const int head  = blockIdx.y;
  const int which = blockIdx.z;               // 0=Q 1=K 2=V
  const int nBase = head * HSZ;
  const float* Wp = (which == 0) ? Wq : ((which == 1) ? Wk : Wv);

  v8f acc[2][4];
#pragma unroll
  for (int m = 0; m < 2; ++m)
#pragma unroll
    for (int j = 0; j < 4; ++j)
#pragma unroll
      for (int i = 0; i < 8; ++i) acc[m][j][i] = 0.f;

  const int r0 = t >> 2;            // 0..31
  const int kc = (t & 3) * 8;       // 0,8,16,24

  for (int kt = 0; kt < HDIM; kt += 32) {
    // batch all global loads first, then convert, then LDS stores
    float4 xg[8], wg[4];
#pragma unroll
    for (int ph = 0; ph < 4; ++ph) {
      const float4* p = (const float4*)(X + (size_t)(mBase + r0 + ph * 32) * HDIM + kt + kc);
      xg[2 * ph] = p[0];
      xg[2 * ph + 1] = p[1];
    }
#pragma unroll
    for (int ph = 0; ph < 2; ++ph) {
      const float4* p = (const float4*)(Wp + (size_t)(nBase + r0 + ph * 32) * HDIM + kt + kc);
      wg[2 * ph] = p[0];
      wg[2 * ph + 1] = p[1];
    }
#pragma unroll
    for (int ph = 0; ph < 4; ++ph)
      *(v8h*)(Xs + (r0 + ph * 32) * LDA + kc) = cvt8(xg[2 * ph], xg[2 * ph + 1]);
#pragma unroll
    for (int ph = 0; ph < 2; ++ph)
      *(v8h*)(Ws + (r0 + ph * 32) * LDA + kc) = cvt8(wg[2 * ph], wg[2 * ph + 1]);
    __syncthreads();

    const v16h aF0 = frag_a(Xs, wave * 32, LDA, 0);
    const v16h aF1 = frag_a(Xs, wave * 32 + 16, LDA, 0);
#pragma unroll
    for (int nt = 0; nt < 4; ++nt) {
      const v16h bF = frag_b(Ws, nt * 16, LDA, 0);
      acc[0][nt] = WMMA_F16(aF0, bF, acc[0][nt]);
      acc[1][nt] = WMMA_F16(aF1, bF, acc[1][nt]);
    }
    __syncthreads();
  }

  // C-layout: lane holds (row = vgpr i + (lane<16?0:8), col = tile*16 + lane&15)
  const int b       = mBase >> 11;            // / SEQ
  const int sB      = mBase & (SEQ - 1);
  const int colLane = lane & 15;
  const int rowOff  = (lane < 16) ? 0 : 8;

  if (which < 2) {
    // RoPE fused: partner element (d +/- 32) sits in tile nt^2, same lane/vgpr.
    h16* dst = ((which == 0) ? Qws : Kws) + (size_t)(b * NHEADS + head) * SEQ * HSZ;
    const float kln = 9.210340371976184f / 32.f;   // ln(10000)/32
#pragma unroll
    for (int nt = 0; nt < 4; ++nt) {
      const int   d    = nt * 16 + colLane;
      const float invf = __expf(-(float)(d & 31) * kln);
      const float sgn  = (d < 32) ? -1.f : 1.f;
#pragma unroll
      for (int m = 0; m < 2; ++m)
#pragma unroll
        for (int i = 0; i < 8; ++i) {
          const int   srow = sB + wave * 32 + m * 16 + rowOff + i;
          const float th   = (float)srow * invf;
          const float v = acc[m][nt][i] * __cosf(th) + sgn * acc[m][nt ^ 2][i] * __sinf(th);
          dst[(size_t)srow * HSZ + d] = (h16)v;
        }
    }
  } else {
    // V: transpose through LDS, store Vt[b][h][d][s] with coalesced b128 rows.
#pragma unroll
    for (int nt = 0; nt < 4; ++nt)
#pragma unroll
      for (int m = 0; m < 2; ++m)
#pragma unroll
        for (int i = 0; i < 8; ++i)
          stage[(wave * 32 + m * 16 + rowOff + i) * 65 + nt * 16 + colLane] = acc[m][nt][i];
    __syncthreads();
    h16* dst     = Vtws + (size_t)(b * NHEADS + head) * HSZ * SEQ;
    const int d  = t >> 1;          // 0..63
    const int j0 = (t & 1) * 64;    // half of the 128 s-rows
#pragma unroll
    for (int c8 = 0; c8 < 8; ++c8) {
      v8h pk;
#pragma unroll
      for (int e = 0; e < 8; ++e) pk[e] = (h16)stage[(j0 + c8 * 8 + e) * 65 + d];
      *(v8h*)(dst + (size_t)d * SEQ + sB + j0 + c8 * 8) = pk;
    }
  }
}

// ---------------------------------------------------------------------------
// Kernel 2: flash attention.  grid = (S/64, NHEADS, B), block = 128.
// Each wave owns 16 query rows; K / Vt tiles stream through double-buffered
// LDS via the Tensor Data Mover (wave 0 issues; TENSORcnt pipelining).
// ctx ws: [B][S][H] f16 (heads re-interleaved for the output GEMM).
// ---------------------------------------------------------------------------
__global__ __launch_bounds__(128)
void attn_kernel(const h16* __restrict__ Qws, const h16* __restrict__ Kws,
                 const h16* __restrict__ Vtws, h16* __restrict__ Cws) {
  __shared__ h16 Ks[2][64 * LDK];
  __shared__ h16 Vs[2][64 * LDK];
  __shared__ h16 Ps[4][16 * LDK];

  const int t    = threadIdx.x;
  const int wave = t >> 5;
  const int lane = t & 31;
  const int sQ   = blockIdx.x * 64;
  const int head = blockIdx.y;
  const int b    = blockIdx.z;
  const size_t bh = (size_t)(b * NHEADS + head);
  const h16* Qp  = Qws + bh * SEQ * HSZ;
  const h16* Kp  = Kws + bh * SEQ * HSZ;
  const h16* Vp  = Vtws + bh * HSZ * SEQ;    // [d][s]
  const int colLane = lane & 15;
  const int rowOff  = (lane < 16) ? 0 : 8;

  // Q fragments (A layout), loaded once.
  v16h qf[2];
#pragma unroll
  for (int ks = 0; ks < 2; ++ks)
    qf[ks] = frag_a(Qp + (size_t)(sQ + wave * 16) * HSZ, 0, HSZ, ks * 32);

  float mrow[8], lrow[8];
  v8f   oacc[4];
#pragma unroll
  for (int i = 0; i < 8; ++i) { mrow[i] = -1e30f; lrow[i] = 0.f; }
#pragma unroll
  for (int j = 0; j < 4; ++j)
#pragma unroll
    for (int i = 0; i < 8; ++i) oacc[j][i] = 0.f;

  const float scale = 0.125f;               // 1/sqrt(64)

#if USE_TDM
  const u32 ksOff[2] = {ldsOffset(&Ks[0][0]), ldsOffset(&Ks[1][0])};
  const u32 vsOff[2] = {ldsOffset(&Vs[0][0]), ldsOffset(&Vs[1][0])};
  if (wave == 0) {                          // prologue: tile 0 -> buffer 0
    tdm_load_2d(ksOff[0], Kp, HSZ, SEQ, HSZ, 64, HSZ);
    tdm_load_2d(vsOff[0], Vp, SEQ, HSZ, 64, HSZ, SEQ);
  }
#else
  const int rld = t >> 1;                   // 0..63
  const int cld = (t & 1) * 32;
#endif

  for (int jt = 0, jtile = 0; jt < SEQ; jt += 64, ++jtile) {
    const int cur = jtile & 1;

#if USE_TDM
    if (wave == 0) {
      if (jt + 64 < SEQ) {                  // issue next tile into other buffer
        tdm_load_2d(ksOff[cur ^ 1], Kp + (size_t)(jt + 64) * HSZ, HSZ, SEQ, HSZ, 64, HSZ);
        tdm_load_2d(vsOff[cur ^ 1], Vp + (jt + 64), SEQ, HSZ, 64, HSZ, SEQ);
        // in-order TDM completion: <=2 outstanding means current pair landed
        __builtin_amdgcn_s_wait_tensorcnt((short)2);
      } else {
        __builtin_amdgcn_s_wait_tensorcnt((short)0);
      }
    }
#else
#pragma unroll
    for (int c8 = 0; c8 < 4; ++c8) {
      *(v8h*)(Ks[cur] + rld * LDK + cld + c8 * 8) =
          *(const v8h*)(Kp + (size_t)(jt + rld) * HSZ + cld + c8 * 8);
      *(v8h*)(Vs[cur] + rld * LDK + cld + c8 * 8) =
          *(const v8h*)(Vp + (size_t)rld * SEQ + jt + cld + c8 * 8);
    }
#endif
    __syncthreads();

    // scores = Q (16x64) x K^T (64x64) per wave
    v8f sc[4];
#pragma unroll
    for (int nt = 0; nt < 4; ++nt) {
#pragma unroll
      for (int i = 0; i < 8; ++i) sc[nt][i] = 0.f;
      sc[nt] = WMMA_F16(qf[0], frag_b(Ks[cur], nt * 16, LDK, 0),  sc[nt]);
      sc[nt] = WMMA_F16(qf[1], frag_b(Ks[cur], nt * 16, LDK, 32), sc[nt]);
    }

    // online softmax; rows live in 16-lane half-groups -> shfl_xor width 16
#pragma unroll
    for (int i = 0; i < 8; ++i) {
      float mx = fmaxf(fmaxf(sc[0][i], sc[1][i]), fmaxf(sc[2][i], sc[3][i])) * scale;
#pragma unroll
      for (int off = 8; off; off >>= 1) mx = fmaxf(mx, __shfl_xor(mx, off, 16));
      const float mnew  = fmaxf(mrow[i], mx);
      const float alpha = __expf(mrow[i] - mnew);
      mrow[i] = mnew;
      float rs = 0.f;
#pragma unroll
      for (int nt = 0; nt < 4; ++nt) {
        const float p = __expf(sc[nt][i] * scale - mnew);
        sc[nt][i] = p;
        rs += p;
      }
#pragma unroll
      for (int off = 8; off; off >>= 1) rs += __shfl_xor(rs, off, 16);
      lrow[i] = lrow[i] * alpha + rs;
#pragma unroll
      for (int nt = 0; nt < 4; ++nt) oacc[nt][i] *= alpha;
    }

    // C-layout -> A-layout for P via this wave's private LDS slab.
#pragma unroll
    for (int nt = 0; nt < 4; ++nt)
#pragma unroll
      for (int i = 0; i < 8; ++i)
        Ps[wave][(rowOff + i) * LDK + nt * 16 + colLane] = (h16)sc[nt][i];
    asm volatile("s_wait_dscnt 0" ::: "memory");   // own-wave LDS RAW fence
    const v16h pf0 = frag_a(Ps[wave], 0, LDK, 0);
    const v16h pf1 = frag_a(Ps[wave], 0, LDK, 32);

    // O += P (16x64) x V (64x64); B-frags read Vt rows (contiguous along j)
#pragma unroll
    for (int dt = 0; dt < 4; ++dt) {
      oacc[dt] = WMMA_F16(pf0, frag_b(Vs[cur], dt * 16, LDK, 0),  oacc[dt]);
      oacc[dt] = WMMA_F16(pf1, frag_b(Vs[cur], dt * 16, LDK, 32), oacc[dt]);
    }
    __syncthreads();
  }

  // Normalize and scatter ctx into [B][S][H] (heads interleaved back).
  h16* dst = Cws + (size_t)b * SEQ * HDIM;
#pragma unroll
  for (int dt = 0; dt < 4; ++dt)
#pragma unroll
    for (int i = 0; i < 8; ++i) {
      const int   srow = sQ + wave * 16 + rowOff + i;
      const float v    = oacc[dt][i] / lrow[i];
      dst[(size_t)srow * HDIM + head * HSZ + dt * 16 + colLane] = (h16)v;
    }
}

// ---------------------------------------------------------------------------
// Kernel 3: out = ctx @ Wo^T.  grid = (B*S/128, H/64), block = 128.
// ---------------------------------------------------------------------------
__global__ __launch_bounds__(128)
void out_proj_kernel(const h16* __restrict__ Cws, const float* __restrict__ Wo,
                     float* __restrict__ Out) {
  __shared__ h16 As[128 * LDA];
  __shared__ h16 Bs[64 * LDA];

  const int t     = threadIdx.x;
  const int wave  = t >> 5;
  const int lane  = t & 31;
  const int mBase = blockIdx.x * 128;
  const int nBase = blockIdx.y * 64;

  v8f acc[2][4];
#pragma unroll
  for (int m = 0; m < 2; ++m)
#pragma unroll
    for (int j = 0; j < 4; ++j)
#pragma unroll
      for (int i = 0; i < 8; ++i) acc[m][j][i] = 0.f;

  const int r0 = t >> 2;
  const int kc = (t & 3) * 8;

  for (int kt = 0; kt < HDIM; kt += 32) {
    v8h ag[4];
    float4 wg[4];
#pragma unroll
    for (int ph = 0; ph < 4; ++ph)
      ag[ph] = *(const v8h*)(Cws + (size_t)(mBase + r0 + ph * 32) * HDIM + kt + kc);
#pragma unroll
    for (int ph = 0; ph < 2; ++ph) {
      const float4* p = (const float4*)(Wo + (size_t)(nBase + r0 + ph * 32) * HDIM + kt + kc);
      wg[2 * ph] = p[0];
      wg[2 * ph + 1] = p[1];
    }
#pragma unroll
    for (int ph = 0; ph < 4; ++ph)
      *(v8h*)(As + (r0 + ph * 32) * LDA + kc) = ag[ph];
#pragma unroll
    for (int ph = 0; ph < 2; ++ph)
      *(v8h*)(Bs + (r0 + ph * 32) * LDA + kc) = cvt8(wg[2 * ph], wg[2 * ph + 1]);
    __syncthreads();

    const v16h aF0 = frag_a(As, wave * 32, LDA, 0);
    const v16h aF1 = frag_a(As, wave * 32 + 16, LDA, 0);
#pragma unroll
    for (int nt = 0; nt < 4; ++nt) {
      const v16h bF = frag_b(Bs, nt * 16, LDA, 0);
      acc[0][nt] = WMMA_F16(aF0, bF, acc[0][nt]);
      acc[1][nt] = WMMA_F16(aF1, bF, acc[1][nt]);
    }
    __syncthreads();
  }

  const int colLane = lane & 15;
  const int rowOff  = (lane < 16) ? 0 : 8;
#pragma unroll
  for (int nt = 0; nt < 4; ++nt)
#pragma unroll
    for (int m = 0; m < 2; ++m)
#pragma unroll
      for (int i = 0; i < 8; ++i)
        Out[(size_t)(mBase + wave * 32 + m * 16 + rowOff + i) * HDIM +
            nBase + nt * 16 + colLane] = acc[m][nt][i];
}

// ---------------------------------------------------------------------------
extern "C" void kernel_launch(void* const* d_in, const int* in_sizes, int n_in,
                              void* d_out, int out_size, void* d_ws, size_t ws_size,
                              hipStream_t stream) {
  (void)in_sizes; (void)n_in; (void)out_size; (void)ws_size;
  const float* X  = (const float*)d_in[0];
  const float* Wq = (const float*)d_in[1];
  const float* Wk = (const float*)d_in[2];
  const float* Wv = (const float*)d_in[3];
  const float* Wo = (const float*)d_in[4];
  float*       Out = (float*)d_out;

  const size_t perTensor = (size_t)NB * NHEADS * SEQ * HSZ;   // 4M halves = 8MB
  h16* Qws  = (h16*)d_ws;
  h16* Kws  = Qws + perTensor;
  h16* Vtws = Kws + perTensor;
  h16* Cws  = Vtws + perTensor;                                // [B][S][H]

  qkv_rope_kernel<<<dim3(NB * SEQ / 128, NHEADS, 3), 128, 0, stream>>>(
      X, Wq, Wk, Wv, Qws, Kws, Vtws);
  attn_kernel<<<dim3(SEQ / 64, NHEADS, NB), 128, 0, stream>>>(
      Qws, Kws, Vtws, Cws);
  out_proj_kernel<<<dim3(NB * SEQ / 128, HDIM / 64), 128, 0, stream>>>(
      Cws, Wo, Out);
}
